// Net_51067161150240
// MI455X (gfx1250) — matrix operand order repaired and checked by reading, not compiled, not run on hardware
//
#include <hip/hip_runtime.h>

// ---------------------------------------------------------------------------
// GCN on MI455X (gfx1250): bf16 WMMA GEMMs + coalesced atomic scatter.
// ---------------------------------------------------------------------------

#define DIN   48
#define DHID  300
#define HP    320     // hidden padded to 10 k-tiles of 32 / 20 n-tiles of 16
#define KPPRE 64      // 48 padded to 2 k-tiles of 32
#define NTILES 20     // HP / 16
#define NGROUPS 5     // n-tile groups per row of tiles
#define NTJ    4      // n-tiles per wave (NGROUPS * NTJ == NTILES)

typedef __attribute__((ext_vector_type(16))) __bf16 v16bf;
typedef __attribute__((ext_vector_type(8)))  float  v8f;

union AFrag { v16bf v; uint4 q[2]; };

static __device__ __forceinline__ __bf16 f2bf(float f) {
  unsigned u = __float_as_uint(f);
  unsigned r = (u + 0x7FFFu + ((u >> 16) & 1u)) >> 16;
  unsigned short h = (unsigned short)r;
  return __builtin_bit_cast(__bf16, h);
}

// ----------------------------- small helpers -------------------------------

__global__ void fill_f32(float* __restrict__ p, float v, long n) {
  long i = (long)blockIdx.x * blockDim.x + threadIdx.x;
  if (i < n) p[i] = v;
}

__global__ void deg_scatter(const int* __restrict__ ei, float* __restrict__ deg, int E) {
  int e = blockIdx.x * blockDim.x + threadIdx.x;
  if (e < E) atomicAdd(&deg[ei[E + e]], 1.0f);   // dst = ei[1][e]
}

__global__ void compute_dinv(const float* __restrict__ deg, float* __restrict__ dinv, int N) {
  int i = blockIdx.x * blockDim.x + threadIdx.x;
  if (i < N) dinv[i] = rsqrtf(deg[i]);
}

__global__ void compute_norm(const int* __restrict__ ei, const float* __restrict__ dinv,
                             float* __restrict__ nrm, int E) {
  int e = blockIdx.x * blockDim.x + threadIdx.x;
  if (e < E) nrm[e] = dinv[ei[e]] * dinv[ei[E + e]];
}

__global__ void cast_x(const float* __restrict__ x, __bf16* __restrict__ xb, long total) {
  long idx = (long)blockIdx.x * blockDim.x + threadIdx.x;
  if (idx >= total) return;
  long i = idx >> 6;        // row
  int  c = (int)(idx & 63); // padded col
  float v = (c < DIN) ? x[i * DIN + c] : 0.0f;
  xb[idx] = f2bf(v);
}

// Pack W [Ksrc x 300] row-major (f32) into per-lane WMMA B-fragment layout:
// P[((kt*NTILES + nt)*32 + lane)*16 + i] = W[kt*32 + 8*(lane>>4) + (i&7) + 16*(i>>3)][nt*16 + (lane&15)]
__global__ void pack_w(const float* __restrict__ W, __bf16* __restrict__ P, int Ksrc, long total) {
  long idx = (long)blockIdx.x * blockDim.x + threadIdx.x;
  if (idx >= total) return;
  int  i    = (int)(idx & 15);
  int  lane = (int)((idx >> 4) & 31);
  long t    = idx >> 9;
  int  nt   = (int)(t % NTILES);
  int  kt   = (int)(t / NTILES);
  int  k = kt * 32 + ((lane >> 4) << 3) + (i & 7) + ((i >> 3) << 4);
  int  n = nt * 16 + (lane & 15);
  float v = (k < Ksrc && n < DHID) ? W[(long)k * DHID + n] : 0.0f;
  P[idx] = f2bf(v);
}

// ------------------------------ WMMA GEMM ----------------------------------
// C[m,n] = sum_k A[m,k] * B[k,n], A row-major bf16 (lda), B pre-packed frags.
// EPI==0: hw = C ; agg = C * dinv[row]^2            (GCN layer)
// EPI==1: hOut = bf16(relu(C + bias[col])), pad->0  (feature_pre)
template<int KT, int EPI>
__global__ void __launch_bounds__(128)
gemm_wmma(const __bf16* __restrict__ A, int lda,
          const __bf16* __restrict__ Bp,
          const float* __restrict__ bias,
          const float* __restrict__ dinv,
          float* __restrict__ hw, float* __restrict__ agg,
          __bf16* __restrict__ hOut,
          int mtiles)
{
  const int lane = threadIdx.x & 31;
  const int wid  = blockIdx.x * 4 + (threadIdx.x >> 5);
  if (wid >= mtiles * NGROUPS) return;           // wave-uniform: EXEC all-ones for WMMA
  const int mtile = wid / NGROUPS;
  const int ng    = wid - mtile * NGROUPS;
  const int m = lane & 15;
  const int g = lane >> 4;

  const __bf16* arow = A + (size_t)(mtile * 16 + m) * lda;

  v8f acc[NTJ] = {};

#pragma unroll
  for (int kt = 0; kt < KT; ++kt) {
    AFrag a;
    a.q[0] = *(const uint4*)(arow + kt * 32 + 8 * g);        // K = 8g .. 8g+7
    a.q[1] = *(const uint4*)(arow + kt * 32 + 16 + 8 * g);   // K = 16+8g .. 23+8g
#pragma unroll
    for (int j = 0; j < NTJ; ++j) {
      const int nt = ng * NTJ + j;
      v16bf b = *(const v16bf*)(Bp + ((size_t)(kt * NTILES + nt) * 32 + lane) * 16);
      acc[j] = __builtin_amdgcn_wmma_f32_16x16x32_bf16(
          false, a.v, false, b, (short)0, acc[j], false, false);
    }
  }

  if (EPI == 0) {
#pragma unroll
    for (int r = 0; r < 8; ++r) {
      const int row = mtile * 16 + r + 8 * g;
      const float s  = dinv[row];
      const float s2 = s * s;
#pragma unroll
      for (int j = 0; j < NTJ; ++j) {
        const int col = (ng * NTJ + j) * 16 + m;
        const size_t idx = (size_t)row * HP + col;
        const float v = acc[j][r];
        hw[idx]  = v;
        agg[idx] = v * s2;                      // self-loop term, fused
      }
    }
  } else {
#pragma unroll
    for (int r = 0; r < 8; ++r) {
      const int row = mtile * 16 + r + 8 * g;
#pragma unroll
      for (int j = 0; j < NTJ; ++j) {
        const int col = (ng * NTJ + j) * 16 + m;
        float o = 0.0f;
        if (col < DHID) o = fmaxf(acc[j][r] + bias[col], 0.0f);
        hOut[(size_t)row * HP + col] = f2bf(o);
      }
    }
  }
}

// ----------------------- graph scatter / pooling ---------------------------

// One block (320 threads) per edge: agg[dst] += norm_e * hw[src], coalesced.
__global__ void __launch_bounds__(HP)
edge_scatter(const int* __restrict__ ei, const float* __restrict__ nrm,
             const float* __restrict__ hw, float* __restrict__ agg, int E)
{
  const int e = blockIdx.x;
  const int c = threadIdx.x;
  const int src = ei[e];
  const int dst = ei[E + e];
  const float w = nrm[e];
  if (c < DHID) {
    float v = w * hw[(size_t)src * HP + c];
    atomicAdd(&agg[(size_t)dst * HP + c], v);
  }
}

__global__ void bias_relu_cast(const float* __restrict__ agg, const float* __restrict__ bias,
                               __bf16* __restrict__ h, long total)
{
  long idx = (long)blockIdx.x * blockDim.x + threadIdx.x;
  if (idx >= total) return;
  int c = (int)(idx % HP);
  float v = (c < DHID) ? fmaxf(agg[idx] + bias[c], 0.0f) : 0.0f;
  h[idx] = f2bf(v);
}

// Final layer: relu(agg + b) accumulated straight into pooled sums.
__global__ void __launch_bounds__(HP)
pool_scatter(const float* __restrict__ agg, const float* __restrict__ bias,
             const int* __restrict__ batch, float* __restrict__ pooled, int N)
{
  const int i = blockIdx.x;
  const int c = threadIdx.x;
  if (c < DHID) {
    float v = fmaxf(agg[(size_t)i * HP + c] + bias[c], 0.0f);
    atomicAdd(&pooled[(size_t)batch[i] * DHID + c], v);
  }
}

__global__ void count_scatter(const int* __restrict__ batch, float* __restrict__ counts, int N) {
  int i = blockIdx.x * blockDim.x + threadIdx.x;
  if (i < N) atomicAdd(&counts[batch[i]], 1.0f);
}

// One wave per graph: out[g] = dot(pooled[g], W_out)/max(count,1) + b_out
__global__ void __launch_bounds__(128)
final_out(const float* __restrict__ pooled, const float* __restrict__ counts,
          const float* __restrict__ Wout, const float* __restrict__ bout,
          float* __restrict__ out, int G)
{
  const int g = blockIdx.x * 4 + (threadIdx.x >> 5);
  const int lane = threadIdx.x & 31;
  if (g >= G) return;
  float s = 0.0f;
  for (int c = lane; c < DHID; c += 32) s += pooled[(size_t)g * DHID + c] * Wout[c];
  for (int off = 16; off > 0; off >>= 1) s += __shfl_down(s, off, 32);
  if (lane == 0) out[g] = s / fmaxf(counts[g], 1.0f) + bout[0];
}

// ------------------------------- launcher ----------------------------------

extern "C" void kernel_launch(void* const* d_in, const int* in_sizes, int n_in,
                              void* d_out, int out_size, void* d_ws, size_t ws_size,
                              hipStream_t stream)
{
  const float* x    = (const float*)d_in[0];
  const int*   ei   = (const int*)d_in[1];
  const int*   batch= (const int*)d_in[2];
  const float* Wpre = (const float*)d_in[3];
  const float* bpre = (const float*)d_in[4];
  const float* Wgcn = (const float*)d_in[5];
  const float* bgcn = (const float*)d_in[6];
  const float* Wout = (const float*)d_in[7];
  const float* bout = (const float*)d_in[8];
  float* out = (float*)d_out;

  const int N = in_sizes[0] / DIN;   // 100000 (multiple of 16)
  const int E = in_sizes[1] / 2;     // 300000
  const int G = out_size;            // 4096

  // workspace carve (256B aligned)
  char* wp = (char*)d_ws;
  auto carve = [&](size_t bytes) { void* p = wp; wp += (bytes + 255) & ~(size_t)255; return p; };
  float*  deg    = (float*) carve((size_t)N * 4);
  float*  dinv   = (float*) carve((size_t)N * 4);
  float*  nrm    = (float*) carve((size_t)E * 4);
  __bf16* xb     = (__bf16*)carve((size_t)N * KPPRE * 2);
  __bf16* hb     = (__bf16*)carve((size_t)N * HP * 2);
  float*  hw     = (float*) carve((size_t)N * HP * 4);
  float*  agg    = (float*) carve((size_t)N * HP * 4);
  __bf16* Pp     = (__bf16*)carve((size_t)2  * NTILES * 32 * 16 * 2);
  __bf16* Pg     = (__bf16*)carve((size_t)10 * NTILES * 32 * 16 * 2);
  float*  pooled = (float*) carve((size_t)G * DHID * 4);
  float*  counts = (float*) carve((size_t)G * 4);

  const int T = 256;
  // degree / norm precompute + zeroing of accumulators (every call)
  fill_f32<<<(N + T - 1) / T, T, 0, stream>>>(deg, 1.0f, N);
  fill_f32<<<((long)G * DHID + T - 1) / T, T, 0, stream>>>(pooled, 0.0f, (long)G * DHID);
  fill_f32<<<(G + T - 1) / T, T, 0, stream>>>(counts, 0.0f, G);
  deg_scatter<<<(E + T - 1) / T, T, 0, stream>>>(ei, deg, E);
  compute_dinv<<<(N + T - 1) / T, T, 0, stream>>>(deg, dinv, N);
  compute_norm<<<(E + T - 1) / T, T, 0, stream>>>(ei, dinv, nrm, E);

  // operand prep
  cast_x<<<(((long)N * KPPRE) + T - 1) / T, T, 0, stream>>>(x, xb, (long)N * KPPRE);
  pack_w<<<((long)2  * NTILES * 512 + T - 1) / T, T, 0, stream>>>(Wpre, Pp, DIN,  (long)2  * NTILES * 512);
  pack_w<<<((long)10 * NTILES * 512 + T - 1) / T, T, 0, stream>>>(Wgcn, Pg, DHID, (long)10 * NTILES * 512);

  const int mtiles  = (N + 15) / 16;
  const int gblocks = (mtiles * NGROUPS + 3) / 4;

  // feature_pre: h = relu(x @ W_pre + b_pre) -> bf16
  gemm_wmma<2, 1><<<gblocks, 128, 0, stream>>>(xb, KPPRE, Pp, bpre, nullptr,
                                               nullptr, nullptr, hb, mtiles);

  for (int layer = 0; layer < 3; ++layer) {
    // hw = h @ W_gcn ; agg = hw * dinv^2 (fused self-loop)
    gemm_wmma<10, 0><<<gblocks, 128, 0, stream>>>(hb, HP, Pg, nullptr, dinv,
                                                  hw, agg, nullptr, mtiles);
    // agg[dst] += norm_e * hw[src]
    edge_scatter<<<E, HP, 0, stream>>>(ei, nrm, hw, agg, E);
    if (layer < 2) {
      bias_relu_cast<<<(((long)N * HP) + T - 1) / T, T, 0, stream>>>(agg, bgcn, hb, (long)N * HP);
    } else {
      pool_scatter<<<N, HP, 0, stream>>>(agg, bgcn, batch, pooled, N);
    }
  }

  count_scatter<<<(N + T - 1) / T, T, 0, stream>>>(batch, counts, N);
  final_out<<<(G + 3) / 4, 128, 0, stream>>>(pooled, counts, Wout, bout, out, G);
}
